// CausalSelfAttention_11424613007989
// MI455X (gfx1250) — matrix operand order repaired and checked by reading, compiled, not run on hardware
//
#include <hip/hip_runtime.h>

#define DEVINL __device__ __forceinline__

typedef __attribute__((ext_vector_type(16))) __bf16 v16bf;
typedef __attribute__((ext_vector_type(8)))  float  v8f;
typedef __attribute__((ext_vector_type(4)))  unsigned int u32x4;
typedef __attribute__((ext_vector_type(4)))  int i32x4;
typedef __attribute__((ext_vector_type(8)))  int i32x8;

struct __align__(16) U4 { unsigned int x, y, z, w; };
union Frag16 { v16bf v; U4 q[2]; };

constexpr int BB = 4, TT = 2048, CC = 2048, HH = 16, DD = 128;
constexpr int N3 = 3 * CC;     // 6144
constexpr int BT = BB * TT;    // 8192
constexpr float SCALE   = 0.08838834764831845f;  // 1/sqrt(128)
constexpr float LN_ROPE = 13.122363377404328f;   // ln(500000)

// workspace offsets, in ushort (bf16-bit) units; total ~201 MB
constexpr size_t S_XBF   = 0;
constexpr size_t S_WQKVT = S_XBF   + (size_t)BT * CC;   // X as bf16        [8192][2048]
constexpr size_t S_WPT   = S_WQKVT + (size_t)N3 * CC;   // Wqkv^T bf16      [6144][2048]
constexpr size_t S_Q     = S_WPT   + (size_t)CC * CC;   // Wproj^T bf16     [2048][2048]
constexpr size_t S_K     = S_Q     + (size_t)BT * CC;   // Q (roped) bf16   [B,H,T,D]
constexpr size_t S_VT    = S_K     + (size_t)BT * CC;   // K (roped) bf16   [B,H,T,D]
constexpr size_t S_Y     = S_VT    + (size_t)BT * CC;   // V^T bf16         [B,H,D,T]
// S_Y: attention out bf16 [B*T][C]

DEVINL unsigned short f2bf(float f) {
  unsigned int u = __float_as_uint(f);
  u += 0x7FFFu + ((u >> 16) & 1u);   // round-to-nearest-even
  return (unsigned short)(u >> 16);
}

DEVINL v8f wmma_bf16(v16bf a, v16bf b, v8f c) {
  return __builtin_amdgcn_wmma_f32_16x16x32_bf16(false, a, false, b, (short)0, c,
                                                 false, false);
}

// --------------------------------------------------------------- TDM 2-D tile load
// D# per CDNA5 ISA. 2-byte elements, 2-D tile,
// groups 2/3 zero (VADDR2/3 pattern for <=2D tensors). Dims/strides in elements.
DEVINL void tdm_load_tile_2d(unsigned lds_addr, const void* gaddr,
                             int tensor_d0, int tensor_d1,
                             int tile_d0, int tile_d1, int stride0) {
  const unsigned long long ga = (unsigned long long)gaddr;
  u32x4 g0;
  g0[0] = 1u;                                     // count=1 (valid), user mode
  g0[1] = lds_addr;                               // LDS byte address of tile
  g0[2] = (unsigned)ga;                           // global_addr[31:0]
  g0[3] = (unsigned)(ga >> 32) | 0x80000000u;     // global_addr[56:32] | type=2
  i32x8 g1;
  g1[0] = (1 << 16);                              // workgroup_mask=0, data_size=1 (2B)
  g1[1] = (tensor_d0 & 0xFFFF) << 16;             // tensor_dim0[15:0]
  g1[2] = ((tensor_d0 >> 16) & 0xFFFF) | ((tensor_d1 & 0xFFFF) << 16);
  g1[3] = ((tensor_d1 >> 16) & 0xFFFF) | ((tile_d0 & 0xFFFF) << 16);
  g1[4] = (tile_d1 & 0xFFFF);                     // tile_dim1 | tile_dim2=0
  g1[5] = stride0;                                // tensor_dim0_stride[31:0]
  g1[6] = 0;                                      // stride0 hi | stride1 lo
  g1[7] = 0;
  i32x4 z = {};
#if defined(__clang_major__) && (__clang_major__ >= 23)
  i32x8 z8 = {};
  __builtin_amdgcn_tensor_load_to_lds(g0, g1, z, z, z8, 0);
#else
  __builtin_amdgcn_tensor_load_to_lds(g0, g1, z, z, 0);
#endif
}

// ---------------------------------------------------------------- converts
__global__ __launch_bounds__(256) void k_cvt(const float* __restrict__ src,
                                             unsigned short* __restrict__ dst, int n) {
  int i = blockIdx.x * 256 + threadIdx.x;
  if (i < n) dst[i] = f2bf(src[i]);
}

// dst[n][k] = bf16(src[k][n]) : coalesced read, transposed write
__global__ __launch_bounds__(256) void k_cvt_t(const float* __restrict__ src,
                                               unsigned short* __restrict__ dst,
                                               int K, int N) {
  int i = blockIdx.x * 256 + threadIdx.x;
  int k = i / N;
  int n = i - k * N;
  dst[(size_t)n * K + k] = f2bf(src[i]);
}

// ------------------------------------------------- shared GEMM core (K = 2048)
// A: [BT,2048] bf16 row-major. Bt: [N][2048] bf16 (B transposed, n-major).
// Block: 256 threads = 8 waves (2 M x 4 N); wave tile 32x64; block tile 64x256.
// A tiles (64x32, 4 KB) are DMA'd into a double-buffered LDS region by the
// Tensor Data Mover; the DMA for step i+1 overlaps the WMMA work of step i
// (same-wave tensor ops complete in order, so TENSORcnt<=1 => step i resident).
DEVINL void gemm_core(const unsigned short* __restrict__ A,
                      const unsigned short* __restrict__ Bt,
                      int mbase, int nbase,
                      unsigned short* As,       // LDS 2 * 64*32 bf16
                      v8f acc[2][4]) {
  const int lane = threadIdx.x & 31;
  const int wave = threadIdx.x >> 5;
  const int wm = wave >> 2, wn = wave & 3;
  const int nlo = lane & 15, hi = lane >> 4;
  const unsigned lds_as = (unsigned)(unsigned long long)(void*)As;  // LDS aperture low32
  constexpr int NSTEP = CC / 32;                  // 64 K-steps
  constexpr unsigned BUFB = 64 * 32 * 2;          // buffer stride in bytes

  const unsigned short* Atile = A + (size_t)mbase * CC;

  if (wave == 0)                                  // prologue: step 0 -> buf 0
    tdm_load_tile_2d(lds_as, Atile, CC, BT, 32, 64, CC);

  for (int i = 0; i < NSTEP; ++i) {
    const int kb = i * 32;
    if (wave == 0) {
      if (i + 1 < NSTEP) {
        // issue DMA for step i+1 into the other buffer, then wait only until
        // step i's tile is resident (<=1 outstanding): DMA overlaps compute.
        tdm_load_tile_2d(lds_as + (unsigned)((i + 1) & 1) * BUFB,
                         Atile + kb + 32, CC, BT, 32, 64, CC);
        __builtin_amdgcn_s_wait_tensorcnt(1);
      } else {
        __builtin_amdgcn_s_wait_tensorcnt(0);
      }
    }
    __syncthreads();                              // buf[i&1] published to all waves

    const unsigned short* Ab = As + (i & 1) * (64 * 32);
    // A-fragments (this code's k order, same as B: K = hi*16..+15, contiguous)
    v16bf af[2];
#pragma unroll
    for (int mt = 0; mt < 2; ++mt) {
      const unsigned short* r = Ab + (wm * 32 + mt * 16 + nlo) * 32;
      Frag16 u;
      u.q[0] = *(const U4*)(r + hi * 16);
      u.q[1] = *(const U4*)(r + hi * 16 + 8);
      af[mt] = u.v;
    }
    // B-fragments straight from pre-transposed global (k = hi*16 + idx, contiguous)
#pragma unroll
    for (int nt = 0; nt < 4; ++nt) {
      const unsigned short* c =
          Bt + (size_t)(nbase + wn * 64 + nt * 16 + nlo) * CC + kb + hi * 16;
      Frag16 u;
      u.q[0] = ((const U4*)c)[0];
      u.q[1] = ((const U4*)c)[1];
#pragma unroll
      for (int mt = 0; mt < 2; ++mt)
        acc[mt][nt] = wmma_bf16(af[mt], u.v, acc[mt][nt]);
    }
    __syncthreads();  // all waves done reading buf[i&1] before it is re-targeted
  }
}

// -------------------------------------------- QKV GEMM with fused RoPE epilogue
__global__ __launch_bounds__(256) void k_gemm_qkv_rope(unsigned short* __restrict__ ws) {
  __shared__ __align__(16) unsigned short As[2 * 64 * 32];
  const unsigned short* Xbf = ws + S_XBF;
  const unsigned short* Wt  = ws + S_WQKVT;
  unsigned short* Qb = ws + S_Q;
  unsigned short* Kb = ws + S_K;
  unsigned short* Vt = ws + S_VT;

  const int mbase = blockIdx.y * 64;
  const int nbase = blockIdx.x * 256;
  v8f acc[2][4] = {};
  gemm_core(Xbf, Wt, mbase, nbase, As, acc);

  const int lane = threadIdx.x & 31;
  const int wave = threadIdx.x >> 5;
  const int wm = wave >> 2, wn = wave & 3;
  const int nlo = lane & 15, hi = lane >> 4;

#pragma unroll
  for (int mt = 0; mt < 2; ++mt) {
#pragma unroll
    for (int nt = 0; nt < 4; ++nt) {
      const int gn = nbase + wn * 64 + nt * 16 + nlo;
      const int sel = gn >> 11;          // 0=q 1=k 2=v (uniform per 16-tile)
      const int cc = gn & (CC - 1);
      const int h = cc >> 7;
      const int d = cc & (DD - 1);
      const float invf = __expf(-(float)(d & ~1) * (LN_ROPE / (float)DD));
#pragma unroll
      for (int v = 0; v < 8; ++v) {
        const int gm = mbase + wm * 32 + mt * 16 + v + 8 * hi;  // C-layout: M=v+8*hi
        const int b = gm >> 11;
        const int t = gm & (TT - 1);
        const float self = acc[mt][nt][v];
        const float partner = __shfl_xor(self, 1, 32);  // channel d^1 lives in lane^1
        float outv;
        if (sel < 2) {
          float sv, cv;
          __sincosf((float)t * invf, &sv, &cv);
          outv = (d & 1) ? (self * cv + partner * sv)   // odd:  x2*c + x1*s
                         : (self * cv - partner * sv);  // even: x1*c - x2*s
        } else {
          outv = self;
        }
        const unsigned short bits = f2bf(outv);
        if (sel == 0)
          Qb[((size_t)(b * HH + h) * TT + t) * DD + d] = bits;
        else if (sel == 1)
          Kb[((size_t)(b * HH + h) * TT + t) * DD + d] = bits;
        else
          Vt[((size_t)(b * HH + h) * DD + d) * TT + t] = bits;  // transposed V
      }
    }
  }
}

// ---------------------------------------------------- flash attention, 1 wave/tile
__global__ __launch_bounds__(32) void k_flash(unsigned short* __restrict__ ws) {
  __shared__ __align__(16) unsigned short Pt[16 * 32];  // P bounce buffer (C->A relayout)

  const unsigned short* Qb = ws + S_Q;
  const unsigned short* Kb = ws + S_K;
  const unsigned short* Vt = ws + S_VT;
  unsigned short* Yb = ws + S_Y;

  const int lane = threadIdx.x & 31;
  const int nlo = lane & 15, hi = lane >> 4;

  const int qt = blockIdx.x & 127;
  const int hh = (blockIdx.x >> 7) & 15;
  const int bb = blockIdx.x >> 11;
  const int qbase = qt << 4;

  const size_t bhT = ((size_t)bb * HH + hh) * TT;
  const size_t bhD = ((size_t)bb * HH + hh) * DD;

  // Q A-fragments (rows qbase..+15, D split into 4 chunks of 32)
  v16bf aq[4];
  {
    const unsigned short* qrow = Qb + (bhT + qbase + nlo) * DD;
#pragma unroll
    for (int dc = 0; dc < 4; ++dc) {
      Frag16 u;
      u.q[0] = *(const U4*)(qrow + dc * 32 + hi * 16);
      u.q[1] = *(const U4*)(qrow + dc * 32 + hi * 16 + 8);
      aq[dc] = u.v;
    }
  }

  v8f o[8] = {};
  float m_r[8], l_r[8];
#pragma unroll
  for (int v = 0; v < 8; ++v) { m_r[v] = -1e30f; l_r[v] = 0.f; }

  const int nkb = (qt >> 1) + 1;  // causal: key blocks of 32 up to this q tile
  for (int jb = 0; jb < nkb; ++jb) {
    const int kb0 = jb * 32;

    // S = Q @ K^T for 32 keys (two 16-key N tiles)
    v8f s0 = {}, s1 = {};
    {
      const unsigned short* k0b = Kb + (bhT + kb0 + nlo) * DD;
#pragma unroll
      for (int dc = 0; dc < 4; ++dc) {
        Frag16 u0, u1;
        const unsigned short* p0 = k0b + dc * 32 + hi * 16;
        const unsigned short* p1 = p0 + 16 * DD;
        u0.q[0] = ((const U4*)p0)[0]; u0.q[1] = ((const U4*)p0)[1];
        u1.q[0] = ((const U4*)p1)[0]; u1.q[1] = ((const U4*)p1)[1];
        s0 = wmma_bf16(aq[dc], u0.v, s0);
        s1 = wmma_bf16(aq[dc], u1.v, s1);
      }
    }

    const int key0 = kb0 + nlo;
    const int key1 = key0 + 16;
    float alpha[8];
#pragma unroll
    for (int v = 0; v < 8; ++v) {
      const int qrow = qbase + v + 8 * hi;
      const float a0 = (key0 <= qrow) ? s0[v] * SCALE : -1e30f;
      const float a1 = (key1 <= qrow) ? s1[v] * SCALE : -1e30f;
      float rm = fmaxf(a0, a1);   // row max across 16 lanes of this half
      rm = fmaxf(rm, __shfl_xor(rm, 1, 32));
      rm = fmaxf(rm, __shfl_xor(rm, 2, 32));
      rm = fmaxf(rm, __shfl_xor(rm, 4, 32));
      rm = fmaxf(rm, __shfl_xor(rm, 8, 32));
      const float mn = fmaxf(m_r[v], rm);
      const float al = __expf(m_r[v] - mn);
      const float e0 = __expf(a0 - mn);
      const float e1 = __expf(a1 - mn);
      float rs = e0 + e1;
      rs += __shfl_xor(rs, 1, 32);
      rs += __shfl_xor(rs, 2, 32);
      rs += __shfl_xor(rs, 4, 32);
      rs += __shfl_xor(rs, 8, 32);
      m_r[v] = mn;
      l_r[v] = l_r[v] * al + rs;
      alpha[v] = al;
      Pt[(v + 8 * hi) * 32 + nlo]      = f2bf(e0);  // P row m, key k=nlo
      Pt[(v + 8 * hi) * 32 + 16 + nlo] = f2bf(e1);  // key k=16+nlo
    }
#pragma unroll
    for (int dt = 0; dt < 8; ++dt)
#pragma unroll
      for (int v = 0; v < 8; ++v) o[dt][v] *= alpha[v];

    __syncthreads();  // LDS stores visible before A-layout reload
    v16bf ap;
    {
      const unsigned short* r = Pt + nlo * 32;
      Frag16 u;
      u.q[0] = *(const U4*)(r + hi * 16);
      u.q[1] = *(const U4*)(r + hi * 16 + 8);
      ap = u.v;
    }
    __syncthreads();  // reload done before next iteration overwrites Pt

    // O += P(16x32) @ V(32x128); V^T layout makes B-frags contiguous along keys
#pragma unroll
    for (int dt = 0; dt < 8; ++dt) {
      const unsigned short* vp = Vt + (bhD + dt * 16 + nlo) * TT + kb0 + hi * 16;
      Frag16 u;
      u.q[0] = ((const U4*)vp)[0];
      u.q[1] = ((const U4*)vp)[1];
      o[dt] = wmma_bf16(ap, u.v, o[dt]);
    }
  }

  float rl[8];
#pragma unroll
  for (int v = 0; v < 8; ++v) rl[v] = 1.f / l_r[v];
#pragma unroll
  for (int dt = 0; dt < 8; ++dt) {
    const int d = hh * DD + dt * 16 + nlo;
#pragma unroll
    for (int v = 0; v < 8; ++v) {
      const int trow = qbase + v + 8 * hi;
      Yb[((size_t)bb * TT + trow) * CC + d] = f2bf(o[dt][v] * rl[v]);
    }
  }
}

// -------------------------------------------------------------- output projection
__global__ __launch_bounds__(256) void k_gemm_proj(const unsigned short* __restrict__ ws,
                                                   float* __restrict__ out) {
  __shared__ __align__(16) unsigned short As[2 * 64 * 32];
  const unsigned short* Yb = ws + S_Y;
  const unsigned short* Wt = ws + S_WPT;

  const int mbase = blockIdx.y * 64;
  const int nbase = blockIdx.x * 256;
  v8f acc[2][4] = {};
  gemm_core(Yb, Wt, mbase, nbase, As, acc);

  const int lane = threadIdx.x & 31;
  const int wave = threadIdx.x >> 5;
  const int wm = wave >> 2, wn = wave & 3;
  const int nlo = lane & 15, hi = lane >> 4;
#pragma unroll
  for (int mt = 0; mt < 2; ++mt)
#pragma unroll
    for (int nt = 0; nt < 4; ++nt) {
      const int gn = nbase + wn * 64 + nt * 16 + nlo;
#pragma unroll
      for (int v = 0; v < 8; ++v) {
        const int gm = mbase + wm * 32 + mt * 16 + v + 8 * hi;
        out[(size_t)gm * CC + gn] = acc[mt][nt][v];
      }
    }
}

// -------------------------------------------------------------------- launcher
extern "C" void kernel_launch(void* const* d_in, const int* in_sizes, int n_in,
                              void* d_out, int out_size, void* d_ws, size_t ws_size,
                              hipStream_t stream) {
  (void)in_sizes; (void)n_in; (void)out_size; (void)ws_size;
  const float* x     = (const float*)d_in[0];  // [4,2048,2048]
  const float* wqkv  = (const float*)d_in[1];  // [2048,6144]
  const float* wproj = (const float*)d_in[2];  // [2048,2048]
  float* out = (float*)d_out;                  // [4,2048,2048]
  unsigned short* ws = (unsigned short*)d_ws;  // needs ~201 MB

  k_cvt<<<(BT * CC) / 256, 256, 0, stream>>>(x, ws + S_XBF, BT * CC);
  k_cvt_t<<<((size_t)N3 * CC) / 256, 256, 0, stream>>>(wqkv, ws + S_WQKVT, CC, N3);
  k_cvt_t<<<((size_t)CC * CC) / 256, 256, 0, stream>>>(wproj, ws + S_WPT, CC, CC);

  k_gemm_qkv_rope<<<dim3(N3 / 256, BT / 64), 256, 0, stream>>>(ws);
  k_flash<<<BB * HH * (TT / 16), 32, 0, stream>>>(ws);
  k_gemm_proj<<<dim3(CC / 256, BT / 64), 256, 0, stream>>>(ws, out);
}
